// GraphAttentionNetwork_266287972500
// MI455X (gfx1250) — matrix operand order repaired and checked by reading, not compiled, not run on hardware
//
#include <hip/hip_runtime.h>
#include <hip/hip_bf16.h>

// ---------------------------------------------------------------------------
// GAT (2-layer) for MI455X / gfx1250, fused flash-style masked softmax.
// Heavy matmuls via v_wmma_f32_16x16x32_bf16 (wave32 WMMA); dominant
// attention kernel stages B-tiles with the Tensor Data Mover into LDS
// (double-buffered, TENSORcnt-tracked) and shares the P-matrix across
// the workgroup through LDS.
// ---------------------------------------------------------------------------

typedef __attribute__((ext_vector_type(16))) __bf16 v16bf;
typedef __attribute__((ext_vector_type(8)))  __bf16 v8bf;
typedef __attribute__((ext_vector_type(8)))  float  v8f;
typedef __attribute__((ext_vector_type(4)))  unsigned int v4u;
typedef __attribute__((ext_vector_type(8)))  int    v8i;
typedef __attribute__((ext_vector_type(4)))  int    v4i;

#define N_NODES 8192
#define ADJW    256      // 8192 / 32 bit-words per adjacency row
#define F0      512
#define F1      256
#define F2      16

// ---------------- WMMA fragment loaders (contiguous k order, k = 16*half + s, same for A and B) -----

// A (16x32 MxK, bf16, row-major source). Lane L: M = L&15.
// lanes 0-15: K = 0..15; lanes 16-31: K = 16..31.
__device__ __forceinline__ v16bf load_a_rowmajor(const __bf16* __restrict__ A,
                                                 int lda, int row, int k0, int laneHi) {
  const __bf16* p = A + (size_t)row * lda + k0 + (laneHi ? 16 : 0);
  v8bf lo = *(const v8bf*)(p);
  v8bf hi = *(const v8bf*)(p + 8);
  v16bf r;
#pragma unroll
  for (int i = 0; i < 8; ++i) { r[i] = lo[i]; r[i + 8] = hi[i]; }
  return r;
}

// B (32x16 KxN, bf16) from a transposed [Ncols][K] row-major buffer:
// lane L: N = L&15; lanes 0-15 hold K base..base+15, lanes 16-31 K base+16..+31.
__device__ __forceinline__ v16bf load_b_transposed(const __bf16* __restrict__ Bt,
                                                   int ldb, int nrow, int kbase) {
  const __bf16* p = Bt + (size_t)nrow * ldb + kbase;
  v8bf lo = *(const v8bf*)(p);
  v8bf hi = *(const v8bf*)(p + 8);
  v16bf r;
#pragma unroll
  for (int i = 0; i < 8; ++i) { r[i] = lo[i]; r[i + 8] = hi[i]; }
  return r;
}

// B fragment from an LDS-resident [Nrows][32] bf16 tile (row stride 64 B).
__device__ __forceinline__ v16bf load_b_lds(const __bf16* tile, int nrow, int laneHi) {
  const __bf16* p = tile + nrow * 32 + (laneHi ? 16 : 0);
  v8bf lo = *(const v8bf*)(p);
  v8bf hi = *(const v8bf*)(p + 8);
  v16bf r;
#pragma unroll
  for (int i = 0; i < 8; ++i) { r[i] = lo[i]; r[i + 8] = hi[i]; }
  return r;
}

// A fragment from an LDS-resident [16][32] bf16 P-tile.
__device__ __forceinline__ v16bf load_a_lds(const __bf16* ptile, int lane, int laneHi) {
  const __bf16* p = ptile + (lane & 15) * 32 + (laneHi ? 16 : 0);
  v8bf lo = *(const v8bf*)(p);
  v8bf hi = *(const v8bf*)(p + 8);
  v16bf r;
#pragma unroll
  for (int i = 0; i < 8; ++i) { r[i] = lo[i]; r[i + 8] = hi[i]; }
  return r;
}

// Attention-probability A-fragment computed in registers (layer-2 path):
__device__ __forceinline__ v16bf make_p_frag(unsigned word, float fs,
                                             const float* __restrict__ fd,
                                             int j0, float m, float inv_s, int laneHi) {
  int kb = laneHi ? 16 : 0;
  v16bf a;
#pragma unroll
  for (int i = 0; i < 8; ++i) {
    {
      int k = kb + i;
      float e = fs + fd[j0 + k];
      e = (e > 0.f) ? e : 0.2f * e;
      float p = ((word >> k) & 1u) ? __expf(e - m) * inv_s : 0.f;
      a[i] = (__bf16)p;
    }
    {
      int k = kb + 8 + i;
      float e = fs + fd[j0 + k];
      e = (e > 0.f) ? e : 0.2f * e;
      float p = ((word >> k) & 1u) ? __expf(e - m) * inv_s : 0.f;
      a[i + 8] = (__bf16)p;
    }
  }
  return a;
}

// ---------------- Tensor Data Mover: 2D bf16 tile (rows x 32) -> LDS --------
// Descriptor per CDNA5 ISA §8 (D# groups). tensor = [nrows][8192] bf16
// row-major; tile = [nrows][32] starting at column j0; dest LDS packed
// [nrows][32] (row stride 64 B).
__device__ __forceinline__ void tdm_load_tile(unsigned lds_off,
                                              const __bf16* __restrict__ gsrc,
                                              unsigned nrows) {
  unsigned long long ga = (unsigned long long)(uintptr_t)gsrc;
  v4u g0;
  g0[0] = 1u;                                        // count=1 (valid), user mode
  g0[1] = lds_off;                                   // lds_addr (bytes)
  g0[2] = (unsigned)(ga & 0xffffffffu);              // global_addr[31:0]
  g0[3] = (unsigned)((ga >> 32) & 0x01ffffffu) | (2u << 30);  // addr[56:32], type=2
  v8i g1;
  g1[0] = 0x00010000;                 // workgroup_mask=0, data_size=1 (2 B)
  g1[1] = (int)(0x2000u << 16);       // tensor_dim0 = 8192 (low 16 bits)
  g1[2] = (int)(nrows << 16);         // tensor_dim0 hi = 0 | tensor_dim1 lo = nrows
  g1[3] = (int)(32u << 16);           // tensor_dim1 hi = 0 | tile_dim0 = 32
  g1[4] = (int)nrows;                 // tile_dim1 = nrows, tile_dim2 = 0
  g1[5] = 8192;                       // tensor_dim0_stride[31:0] = 8192 elements
  g1[6] = 0;                          // stride hi / tensor_dim1_stride lo
  g1[7] = 0;
  v4i z4 = {0, 0, 0, 0};
#if __clang_major__ >= 23
  v8i z8 = {0, 0, 0, 0, 0, 0, 0, 0};
  __builtin_amdgcn_tensor_load_to_lds(g0, g1, z4, z4, z8, 0);
#else
  __builtin_amdgcn_tensor_load_to_lds(g0, g1, z4, z4, 0);
#endif
}

// ---------------- setup kernels --------------------------------------------

__global__ void k_clear(unsigned* __restrict__ p, int n) {
  int i = blockIdx.x * blockDim.x + threadIdx.x;
  if (i < n) p[i] = 0u;
}

__global__ void k_scatter(const int* __restrict__ ei, int E, unsigned* __restrict__ adj) {
  int e = blockIdx.x * blockDim.x + threadIdx.x;
  if (e >= E) return;
  int s = ei[e];
  int d = ei[E + e];
  atomicOr(adj + (size_t)s * ADJW + (d >> 5), 1u << (d & 31));
}

__global__ void k_cvt_bf16(const float* __restrict__ x, __bf16* __restrict__ y, int n) {
  int i = blockIdx.x * blockDim.x + threadIdx.x;
  if (i < n) y[i] = (__bf16)x[i];
}

// W1 [F0][F1] row-major -> W1t [F1][F0] bf16 (B-operand friendly)
__global__ void k_packW1t(const float* __restrict__ W1, __bf16* __restrict__ W1t) {
  int idx = blockIdx.x * blockDim.x + threadIdx.x;
  if (idx >= F0 * F1) return;
  int n = idx & (F1 - 1);
  int k = idx >> 8;
  W1t[(size_t)n * F0 + k] = (__bf16)W1[idx];
}

// ---------------- GEMM1: h1 = x @ W1 (bf16 WMMA), writes h1^T bf16 ---------

__global__ __launch_bounds__(32) void k_gemm1(const __bf16* __restrict__ xb,
                                              const __bf16* __restrict__ W1t,
                                              __bf16* __restrict__ h1t) {
  int lane = threadIdx.x;
  int laneHi = lane >> 4;
  int i0 = blockIdx.x * 16;
  int c0 = blockIdx.y * 64;
  int row = i0 + (lane & 15);
  v8f acc[4] = {};
  for (int kc = 0; kc < F0; kc += 32) {
    v16bf a = load_a_rowmajor(xb, F0, row, kc, laneHi);
#pragma unroll
    for (int t = 0; t < 4; ++t) {
      v16bf b = load_b_transposed(W1t, F0, c0 + t * 16 + (lane & 15),
                                  kc + (laneHi ? 16 : 0));
      acc[t] = __builtin_amdgcn_wmma_f32_16x16x32_bf16(false, a, false, b,
                                                       (short)0, acc[t], false, false);
    }
  }
#pragma unroll
  for (int t = 0; t < 4; ++t) {
#pragma unroll
    for (int r = 0; r < 8; ++r) {
      int mrow = (laneHi ? 8 : 0) + r;
      int col = c0 + t * 16 + (lane & 15);
      h1t[(size_t)col * N_NODES + (i0 + mrow)] = (__bf16)acc[t][r];
    }
  }
}

// ---------------- attention scalars f_src/f_dst ----------------------------

__global__ void k_fdots1(const __bf16* __restrict__ h1t,
                         const float* __restrict__ a1s, const float* __restrict__ a1d,
                         float* __restrict__ f1s, float* __restrict__ f1d) {
  int i = blockIdx.x * blockDim.x + threadIdx.x;
  if (i >= N_NODES) return;
  float s = 0.f, d = 0.f;
  for (int c = 0; c < F1; ++c) {
    float h = (float)h1t[(size_t)c * N_NODES + i];
    s += h * a1s[c];
    d += h * a1d[c];
  }
  f1s[i] = s;
  f1d[i] = d;
}

// ---------------- per-row online softmax stats (one wave per row) ----------

__global__ __launch_bounds__(32) void k_stats(const unsigned* __restrict__ adj,
                                              const float* __restrict__ fsrc,
                                              const float* __restrict__ fdst,
                                              float* __restrict__ mrow,
                                              float* __restrict__ srow) {
  int row = blockIdx.x;
  int lane = threadIdx.x;
  float fs = fsrc[row];
  const unsigned* aw = adj + (size_t)row * ADJW;
  float m = -3.0e38f, s = 0.f;
  for (int w = 0; w < ADJW; ++w) {
    unsigned word = aw[w];
    if ((word >> lane) & 1u) {
      float e = fs + fdst[w * 32 + lane];
      e = (e > 0.f) ? e : 0.2f * e;
      float mn = fmaxf(m, e);
      s = s * __expf(m - mn) + __expf(e - mn);
      m = mn;
    }
  }
#pragma unroll
  for (int off = 16; off > 0; off >>= 1) {
    float om = __shfl_xor(m, off, 32);
    float os = __shfl_xor(s, off, 32);
    float mn = fmaxf(m, om);
    s = s * __expf(m - mn) + os * __expf(om - mn);
    m = mn;
  }
  if (lane == 0) { mrow[row] = m; srow[row] = s; }
}

// ---------------- fused masked-softmax attention, layer 1 ------------------
// One 256-thread workgroup per 16-row block, full 256-col output.
// Per 32-col chunk: TDM DMAs the 256x32 bf16 B-tile into LDS (double
// buffered, TENSORcnt), all threads cooperatively build the 16x32 P-tile
// in LDS once, then 8 waves run 2 WMMAs each from LDS fragments.

__global__ __launch_bounds__(256) void k_attn1(const unsigned* __restrict__ adj,
                                               const float* __restrict__ f1s,
                                               const float* __restrict__ f1d,
                                               const float* __restrict__ m1,
                                               const float* __restrict__ s1,
                                               const __bf16* __restrict__ h1t,
                                               float* __restrict__ g1) {
  __shared__ __bf16 Hbuf[2][F1 * 32];   // 2 x 16 KB B-tiles
  __shared__ __bf16 Pbuf[2][16 * 32];   // 2 x 1 KB  P-tiles

  int tid = threadIdx.x;
  int lane = tid & 31;
  int wave = tid >> 5;
  int laneHi = (lane >> 4) & 1;
  int i0 = blockIdx.x * 16;

  // P-producer mapping: thread -> (row, k-pair)
  int prow = tid >> 4;           // 0..15
  int pk = (tid & 15) * 2;       // 0,2,...,30
  float fs = f1s[i0 + prow];
  float m = m1[i0 + prow];
  float inv_s = 1.0f / s1[i0 + prow];
  const unsigned* adjrow = adj + (size_t)(i0 + prow) * ADJW;

  // WMMA consumer mapping: wave handles n-tiles 2*wave, 2*wave+1
  int n0 = (wave * 2) * 16 + (lane & 15);
  int n1 = (wave * 2 + 1) * 16 + (lane & 15);
  v8f acc0 = {}, acc1 = {};

  unsigned hoff0 = (unsigned)(uintptr_t)(&Hbuf[0][0]);
  unsigned hoff1 = (unsigned)(uintptr_t)(&Hbuf[1][0]);

  // Cooperative P-tile producer for chunk jc into Pbuf[dst].
  auto produce_p = [&](int jc, int dst) {
    unsigned word = adjrow[jc];
    int j0 = jc * 32;
    float fd0 = f1d[j0 + pk];
    float fd1 = f1d[j0 + pk + 1];
    float e0 = fs + fd0; e0 = (e0 > 0.f) ? e0 : 0.2f * e0;
    float e1 = fs + fd1; e1 = (e1 > 0.f) ? e1 : 0.2f * e1;
    float p0 = ((word >> pk) & 1u) ? __expf(e0 - m) * inv_s : 0.f;
    float p1 = ((word >> (pk + 1)) & 1u) ? __expf(e1 - m) * inv_s : 0.f;
    union { __bf16 h[2]; unsigned u; } pr;
    pr.h[0] = (__bf16)p0;
    pr.h[1] = (__bf16)p1;
    *(unsigned*)&Pbuf[dst][prow * 32 + pk] = pr.u;
  };

  // ---- pipeline prologue: chunk 0 ----
  if (tid < 32) tdm_load_tile(hoff0, h1t + 0, F1);
  produce_p(0, 0);
  if (tid < 32) __builtin_amdgcn_s_wait_tensorcnt(0);
  __syncthreads();

  for (int jc = 0; jc < ADJW; ++jc) {
    int idx = jc & 1;
    // prefetch next chunk: TDM B-tile + cooperative P-tile
    if (jc + 1 < ADJW) {
      if (tid < 32)
        tdm_load_tile(idx ? hoff0 : hoff1, h1t + (jc + 1) * 32, F1);
      produce_p(jc + 1, idx ^ 1);
    }
    // consume current chunk from LDS
    v16bf a = load_a_lds(&Pbuf[idx][0], lane, laneHi);
    v16bf b0 = load_b_lds(&Hbuf[idx][0], n0, laneHi);
    acc0 = __builtin_amdgcn_wmma_f32_16x16x32_bf16(false, a, false, b0,
                                                   (short)0, acc0, false, false);
    v16bf b1 = load_b_lds(&Hbuf[idx][0], n1, laneHi);
    acc1 = __builtin_amdgcn_wmma_f32_16x16x32_bf16(false, a, false, b1,
                                                   (short)0, acc1, false, false);
    // next tile landed before anyone proceeds; reads of buffers[idx] done
    if (tid < 32 && jc + 1 < ADJW) __builtin_amdgcn_s_wait_tensorcnt(0);
    __syncthreads();
  }

  // epilogue: ELU, store g1 (fp32 row-major)
#pragma unroll
  for (int r = 0; r < 8; ++r) {
    int mrow = (laneHi ? 8 : 0) + r;
    float v0 = acc0[r];
    v0 = (v0 > 0.f) ? v0 : (__expf(v0) - 1.0f);
    g1[(size_t)(i0 + mrow) * F1 + n0] = v0;
    float v1 = acc1[r];
    v1 = (v1 > 0.f) ? v1 : (__expf(v1) - 1.0f);
    g1[(size_t)(i0 + mrow) * F1 + n1] = v1;
  }
}

// ---------------- layer 2: h2 = g1 @ W2 (tiny, fp32 scalar) ----------------

__global__ void k_h2(const float* __restrict__ g1, const float* __restrict__ W2,
                     float* __restrict__ h2f, __bf16* __restrict__ h2t) {
  int idx = blockIdx.x * blockDim.x + threadIdx.x;
  if (idx >= N_NODES * F2) return;
  int i = idx >> 4;
  int c = idx & 15;
  float acc = 0.f;
  for (int k = 0; k < F1; ++k)
    acc += g1[(size_t)i * F1 + k] * W2[k * F2 + c];
  h2f[idx] = acc;
  h2t[(size_t)c * N_NODES + i] = (__bf16)acc;
}

__global__ void k_fdots2(const float* __restrict__ h2f,
                         const float* __restrict__ a2s, const float* __restrict__ a2d,
                         float* __restrict__ f2s, float* __restrict__ f2d) {
  int i = blockIdx.x * blockDim.x + threadIdx.x;
  if (i >= N_NODES) return;
  float s = 0.f, d = 0.f;
#pragma unroll
  for (int c = 0; c < F2; ++c) {
    float h = h2f[i * F2 + c];
    s += h * a2s[c];
    d += h * a2d[c];
  }
  f2s[i] = s;
  f2d[i] = d;
}

// ---------------- fused attention layer 2 + log_softmax (WMMA) -------------

__global__ __launch_bounds__(32) void k_attn2(const unsigned* __restrict__ adj,
                                              const float* __restrict__ f2s,
                                              const float* __restrict__ f2d,
                                              const float* __restrict__ m2,
                                              const float* __restrict__ s2,
                                              const __bf16* __restrict__ h2t,
                                              float* __restrict__ out) {
  int lane = threadIdx.x;
  int laneHi = lane >> 4;
  int i0 = blockIdx.x * 16;
  int row = i0 + (lane & 15);
  float fs = f2s[row];
  float m = m2[row];
  float inv_s = 1.0f / s2[row];
  v8f acc = {};
  for (int jc = 0; jc < ADJW; ++jc) {
    unsigned word = adj[(size_t)row * ADJW + jc];
    if (__ballot(word != 0u) == 0ull) continue;   // wave-uniform skip
    int j0 = jc * 32;
    v16bf a = make_p_frag(word, fs, f2d, j0, m, inv_s, laneHi);
    v16bf b = load_b_transposed(h2t, N_NODES, lane & 15, j0 + (laneHi ? 16 : 0));
    acc = __builtin_amdgcn_wmma_f32_16x16x32_bf16(false, a, false, b,
                                                  (short)0, acc, false, false);
  }
  // log_softmax over 16 classes; class index = lane&15 within each 16-lane half
#pragma unroll
  for (int r = 0; r < 8; ++r) {
    float v = acc[r];
    float mx = v;
#pragma unroll
    for (int off = 8; off > 0; off >>= 1)
      mx = fmaxf(mx, __shfl_xor(mx, off, 16));
    float ex = __expf(v - mx);
    float sum = ex;
#pragma unroll
    for (int off = 8; off > 0; off >>= 1)
      sum += __shfl_xor(sum, off, 16);
    float lp = (v - mx) - __logf(sum);
    int mrow = (laneHi ? 8 : 0) + r;
    out[(size_t)(i0 + mrow) * F2 + (lane & 15)] = lp;
  }
}

// ---------------- launch ----------------------------------------------------

extern "C" void kernel_launch(void* const* d_in, const int* in_sizes, int n_in,
                              void* d_out, int out_size, void* d_ws, size_t ws_size,
                              hipStream_t stream) {
  (void)n_in; (void)out_size; (void)ws_size;
  const float* x   = (const float*)d_in[0];
  const int*   ei  = (const int*)d_in[1];
  const float* W1  = (const float*)d_in[2];
  const float* a1s = (const float*)d_in[3];
  const float* a1d = (const float*)d_in[4];
  const float* W2  = (const float*)d_in[5];
  const float* a2s = (const float*)d_in[6];
  const float* a2d = (const float*)d_in[7];
  float* out = (float*)d_out;

  char* ws = (char*)d_ws;
  size_t o = 0;
  auto take = [&](size_t bytes) -> void* {
    void* p = ws + o;
    o += (bytes + 255) & ~(size_t)255;
    return p;
  };
  unsigned* adj = (unsigned*)take((size_t)N_NODES * ADJW * 4);   // 8 MB bitmask
  __bf16* xb    = (__bf16*) take((size_t)N_NODES * F0 * 2);      // 8 MB
  __bf16* W1t   = (__bf16*) take((size_t)F0 * F1 * 2);           // 256 KB
  __bf16* h1t   = (__bf16*) take((size_t)N_NODES * F1 * 2);      // 4 MB (transposed)
  float* f1s    = (float*)  take((size_t)N_NODES * 4);
  float* f1d    = (float*)  take((size_t)N_NODES * 4);
  float* m1     = (float*)  take((size_t)N_NODES * 4);
  float* s1     = (float*)  take((size_t)N_NODES * 4);
  float* g1     = (float*)  take((size_t)N_NODES * F1 * 4);      // 8 MB
  float* h2f    = (float*)  take((size_t)N_NODES * F2 * 4);      // 512 KB
  __bf16* h2t   = (__bf16*) take((size_t)N_NODES * F2 * 2);      // 256 KB (transposed)
  float* f2s    = (float*)  take((size_t)N_NODES * 4);
  float* f2d    = (float*)  take((size_t)N_NODES * 4);
  float* m2     = (float*)  take((size_t)N_NODES * 4);
  float* s2     = (float*)  take((size_t)N_NODES * 4);

  int E = in_sizes[1] / 2;

  k_clear<<<(N_NODES * ADJW + 255) / 256, 256, 0, stream>>>(adj, N_NODES * ADJW);
  k_scatter<<<(E + 255) / 256, 256, 0, stream>>>(ei, E, adj);
  k_cvt_bf16<<<((N_NODES * F0) + 255) / 256, 256, 0, stream>>>(x, xb, N_NODES * F0);
  k_packW1t<<<((F0 * F1) + 255) / 256, 256, 0, stream>>>(W1, W1t);

  k_gemm1<<<dim3(N_NODES / 16, F1 / 64), 32, 0, stream>>>(xb, W1t, h1t);
  k_fdots1<<<(N_NODES + 255) / 256, 256, 0, stream>>>(h1t, a1s, a1d, f1s, f1d);
  k_stats<<<N_NODES, 32, 0, stream>>>(adj, f1s, f1d, m1, s1);
  k_attn1<<<N_NODES / 16, 256, 0, stream>>>(adj, f1s, f1d, m1, s1, h1t, g1);

  k_h2<<<(N_NODES * F2 + 255) / 256, 256, 0, stream>>>(g1, W2, h2f, h2t);
  k_fdots2<<<(N_NODES + 255) / 256, 256, 0, stream>>>(h2f, a2s, a2d, f2s, f2d);
  k_stats<<<N_NODES, 32, 0, stream>>>(adj, f2s, f2d, m2, s2);
  k_attn2<<<N_NODES / 16, 32, 0, stream>>>(adj, f2s, f2d, m2, s2, h2t, out);
}
